// LatentAttention_57329223467464
// MI455X (gfx1250) — matrix operand order repaired and checked by reading, not compiled, not run on hardware
//
#include <hip/hip_runtime.h>
#include <hip/hip_bf16.h>

#define NH 16

typedef __attribute__((ext_vector_type(16))) __bf16          v16bf;
typedef __attribute__((ext_vector_type(16))) unsigned short  v16u;
typedef __attribute__((ext_vector_type(8)))  float           v8f;

static __device__ __forceinline__ v16bf as_bf16x16(v16u u) {
  union { v16u u; v16bf b; } c; c.u = u; return c.b;
}

// fp32 -> bf16 round-to-nearest-even (raw u16)
static __device__ __forceinline__ unsigned short f2bf(float f) {
  union { float f; unsigned int u; } v; v.f = f;
  unsigned int r = v.u + 0x7FFFu + ((v.u >> 16) & 1u);
  return (unsigned short)(r >> 16);
}

__global__ void cvt_f32_bf16(const float* __restrict__ in,
                             unsigned short* __restrict__ out, int n) {
  int stride = gridDim.x * blockDim.x;
  for (int i = blockIdx.x * blockDim.x + threadIdx.x; i < n; i += stride)
    out[i] = f2bf(in[i]);
}

// fp32 [K,N] -> bf16 transposed [N,K]
__global__ void cvt_f32_bf16_t(const float* __restrict__ in,
                               unsigned short* __restrict__ out, int K, int N) {
  int total = K * N;
  int stride = gridDim.x * blockDim.x;
  for (int i = blockIdx.x * blockDim.x + threadIdx.x; i < total; i += stride) {
    int k = i / N, n = i - k * N;
    out[(size_t)n * K + k] = f2bf(in[i]);
  }
}

// int mask -> additive float bias (0 or -1e30)
__global__ void mask_bias(const int* __restrict__ m, float* __restrict__ mb, int n) {
  int stride = gridDim.x * blockDim.x;
  for (int i = blockIdx.x * blockDim.x + threadIdx.x; i < n; i += stride)
    mb[i] = (m[i] != 0) ? 0.0f : -1e30f;
}

// C = A[M,K] @ B (B given transposed as Bt[N,K]), bf16 in, fp32 accumulate,
// bf16 out. Each wave: 32Mx64N tile (2 A-frags x 4 B-frags -> 8 WMMAs per
// 32-wide K step). Block = 4 waves -> 64Mx128N block tile.
// STORE_T=true writes C transposed as Ct[N,M].
template <bool STORE_T>
__global__ void gemm_bf16(const unsigned short* __restrict__ A,
                          const unsigned short* __restrict__ Bt,
                          unsigned short* __restrict__ C,
                          int M, int N, int K) {
  const int wave = threadIdx.x >> 5;
  const int lane = threadIdx.x & 31;
  const int half = lane >> 4;
  const int mn   = lane & 15;
  const int mb   = blockIdx.y * 64 + (wave >> 1) * 32;
  const int nb   = blockIdx.x * 128 + (wave & 1) * 64;

  const unsigned short* Arow[2];
  Arow[0] = A + (size_t)(mb + mn) * K;
  Arow[1] = A + (size_t)(mb + 16 + mn) * K;
  const unsigned short* Brow[4];
#pragma unroll
  for (int f = 0; f < 4; ++f) Brow[f] = Bt + (size_t)(nb + f * 16 + mn) * K;

  v8f acc[2][4] = {};
  for (int kc = 0; kc < K; kc += 32) {
    v16u au[2], bu[4];
#pragma unroll
    for (int i = 0; i < 8; ++i) {
      int ka = half * 16 + 2 * i;
      int kb = half * 16 + 2 * i;
#pragma unroll
      for (int s = 0; s < 2; ++s) {
        unsigned int p = *(const unsigned int*)(Arow[s] + kc + ka);
        au[s][2 * i]     = (unsigned short)(p & 0xFFFFu);
        au[s][2 * i + 1] = (unsigned short)(p >> 16);
      }
#pragma unroll
      for (int f = 0; f < 4; ++f) {
        unsigned int p = *(const unsigned int*)(Brow[f] + kc + kb);
        bu[f][2 * i]     = (unsigned short)(p & 0xFFFFu);
        bu[f][2 * i + 1] = (unsigned short)(p >> 16);
      }
    }
#pragma unroll
    for (int s = 0; s < 2; ++s)
#pragma unroll
      for (int f = 0; f < 4; ++f)
        acc[s][f] = __builtin_amdgcn_wmma_f32_16x16x32_bf16(
            false, as_bf16x16(au[s]), false, as_bf16x16(bu[f]),
            (short)0, acc[s][f], false, false);
  }

#pragma unroll
  for (int s = 0; s < 2; ++s)
#pragma unroll
    for (int f = 0; f < 4; ++f) {
      int col = nb + f * 16 + mn;
      if (STORE_T) {
#pragma unroll
        for (int r = 0; r < 8; r += 2) {
          unsigned int pk = (unsigned int)f2bf(acc[s][f][r]) |
                            ((unsigned int)f2bf(acc[s][f][r + 1]) << 16);
          size_t idx = (size_t)col * M + (mb + s * 16 + 8 * half + r);
          *(unsigned int*)(C + idx) = pk;
        }
      } else {
#pragma unroll
        for (int r = 0; r < 8; ++r) {
          int row = mb + s * 16 + r + 8 * half;
          C[(size_t)row * N + col] = f2bf(acc[s][f][r]);
        }
      }
    }
}

// Causal flash attention, transposed formulation, 32 keys/iteration.
//   S^T = K_tile @ Q^T  (lane = query column; per-lane softmax state)
//   out^T = V^T_tile @ P^T
// K tiles are software-pipelined global->LDS via global_load_async_to_lds_b128
// (ASYNCcnt): tile j0+32 is issued right after the P-store DScnt drain (all
// LDS readers of tile j0 are done by then), and runs in the shadow of the
// P-fragment reads, V loads and P@V WMMAs. Only the loop head waits.
#define PSTRIDE 34  // P-tile LDS row stride (u16): odd bank increment
#define KSTRIDE 72  // K-tile LDS row stride (u16): 144B rows, 16B aligned
struct CFalse { static constexpr bool value = false; };
struct CTrue  { static constexpr bool value = true;  };

__global__ void mla_attn(const unsigned short* __restrict__ Q,
                         const unsigned short* __restrict__ Km,
                         const unsigned short* __restrict__ Vt,
                         const float* __restrict__ mbias,
                         float* __restrict__ out,
                         int T, int D, int BT) {
  __shared__ unsigned short plds[4][16 * PSTRIDE + 2];
  __shared__ unsigned short kls[4][32 * KSTRIDE];

  const int wave = threadIdx.x >> 5;
  const int lane = threadIdx.x & 31;
  const int half = lane >> 4;
  const int mn   = lane & 15;   // this lane's query column
  const int ntq  = T >> 4;

  int w  = blockIdx.x * 4 + wave;
  int qt = w % ntq;
  int h  = (w / ntq) % NH;
  int b  = w / (ntq * NH);
  int bT = b * T;

  // Q^T B-fragments: b[k][n] = Q[query n][dim k] -> contiguous dim pairs
  const unsigned short* qrow = Q + (size_t)(bT + qt * 16 + mn) * D + h * 64;
  v16u qf[2];
#pragma unroll
  for (int c = 0; c < 2; ++c)
#pragma unroll
    for (int i = 0; i < 8; ++i) {
      int kb = half * 16 + 2 * i;
      unsigned int p = *(const unsigned int*)(qrow + c * 32 + kb);
      qf[c][2 * i]     = (unsigned short)(p & 0xFFFFu);
      qf[c][2 * i + 1] = (unsigned short)(p >> 16);
    }

  // Per-lane constant bases (strength-reduced addressing)
  const int lr = lane >> 3, ch = lane & 7;  // async-copy row/chunk per lane
  const unsigned short* gk =                 // K-tile global base for async
      Km + (size_t)bT * D + h * 64 + (size_t)lr * D + ch * 8;
  const unsigned la0 =
      (unsigned)(size_t)&kls[wave][0] + (unsigned)(lr * KSTRIDE * 2 + ch * 16);
  const unsigned short* vrow[4];
#pragma unroll
  for (int f = 0; f < 4; ++f)
    vrow[f] = Vt + (size_t)(h * 64 + f * 16 + mn) * BT + bT;
  const float* mrow = mbias + bT + 8 * half;

  v8f acc[4] = {{}, {}, {}, {}};   // out^T: VGPR r = dim, lane = query
  float rm = -1e30f, rl = 0.0f;    // per-lane (per-query) softmax state
  const int qpos = qt * 16 + mn;
  const int limit = (qt + 1) * 16;  // causal key bound

  // Issue the async global->LDS copy of K tile [j0..j0+31] x 64 dims.
  auto issueK = [&](int j0) {
    const unsigned short* gp = gk + (size_t)j0 * D;
#pragma unroll
    for (int t = 0; t < 8; ++t) {
      unsigned long long ga = (unsigned long long)(size_t)(gp + (size_t)(t * 4) * D);
      unsigned la = la0 + (unsigned)(t * 4 * KSTRIDE * 2);
      asm volatile("global_load_async_to_lds_b128 %0, %1, off"
                   :: "v"(la), "v"(ga) : "memory");
    }
  };

  auto step = [&](auto ctag, int j0) {
    constexpr bool CAUSAL = decltype(ctag)::value;

    // ---- wait for the pipelined K tile ----
    __builtin_amdgcn_wave_barrier();
    asm volatile("s_wait_asynccnt 0" ::: "memory");
    __builtin_amdgcn_wave_barrier();

    // ---- S^T sub-tiles: rows = keys (r + 8*half), cols = queries ----
    v8f st[2] = {{}, {}};
#pragma unroll
    for (int e = 0; e < 2; ++e) {
      const unsigned short* krow = &kls[wave][(e * 16 + mn) * KSTRIDE];
#pragma unroll
      for (int c = 0; c < 2; ++c) {
        v16u kf;
#pragma unroll
        for (int i = 0; i < 8; ++i) {
          int ka = half * 16 + 2 * i;
          unsigned int p = *(const unsigned int*)(krow + c * 32 + ka);
          kf[2 * i]     = (unsigned short)(p & 0xFFFFu);
          kf[2 * i + 1] = (unsigned short)(p >> 16);
        }
        st[e] = __builtin_amdgcn_wmma_f32_16x16x32_bf16(
            false, as_bf16x16(kf), false, as_bf16x16(qf[c]),
            (short)0, st[e], false, false);
      }
    }
    st[0] = st[0] * 0.125f;  // 1/sqrt(64)
    st[1] = st[1] * 0.125f;

    // ---- mask bias (+ causal only on diagonal tile), partial max ----
    float sv[2][8];
    float mx = -1e30f;
#pragma unroll
    for (int e = 0; e < 2; ++e) {
      const float* mp = mrow + j0 + e * 16;
      float4 m0 = *(const float4*)(mp);
      float4 m1 = *(const float4*)(mp + 4);
      float mv[8] = {m0.x, m0.y, m0.z, m0.w, m1.x, m1.y, m1.z, m1.w};
#pragma unroll
      for (int r = 0; r < 8; ++r) {
        float v = st[e][r] + mv[r];
        if (CAUSAL) {
          int kpos = j0 + e * 16 + 8 * half + r;
          v = (kpos <= qpos) ? v : -1e30f;
        }
        sv[e][r] = v;
        mx = fmaxf(mx, v);
      }
    }
    mx = fmaxf(mx, __shfl_xor(mx, 16, 32));  // combine the two key-halves
    float mnew = fmaxf(rm, mx);
    float c0   = __expf(rm - mnew);

    // ---- exp, per-lane partial sum, P^T -> LDS as P[query][key] ----
    float ps = 0.0f;
#pragma unroll
    for (int e = 0; e < 2; ++e) {
      int krel = e * 16 + 8 * half;
#pragma unroll
      for (int r = 0; r < 8; r += 2) {
        float p0 = __expf(sv[e][r] - mnew);
        float p1 = __expf(sv[e][r + 1] - mnew);
        ps += p0 + p1;
        unsigned int pk = (unsigned int)f2bf(p0) | ((unsigned int)f2bf(p1) << 16);
        *(unsigned int*)&plds[wave][mn * PSTRIDE + krel + r] = pk;
      }
    }
    ps += __shfl_xor(ps, 16, 32);
    rl = rl * c0 + ps;
    rm = mnew;
#pragma unroll
    for (int f = 0; f < 4; ++f) acc[f] = acc[f] * c0;  // per-lane rescale

    // Drain DScnt: P-store RAW, and all K-tile LDS readers are now complete.
    __builtin_amdgcn_wave_barrier();
    asm volatile("s_wait_dscnt 0" ::: "memory");
    __builtin_amdgcn_wave_barrier();

    // ---- pipeline: launch next K tile while we finish this one ----
    if (j0 + 32 < limit) issueK(j0 + 32);

    // ---- P^T B-fragment: b[key kb][query n] = plds[n][kb] ----
    v16u pb;
#pragma unroll
    for (int i = 0; i < 8; ++i) {
      int kb = half * 16 + 2 * i;
      unsigned int p = *(const unsigned int*)&plds[wave][mn * PSTRIDE + kb];
      pb[2 * i]     = (unsigned short)(p & 0xFFFFu);
      pb[2 * i + 1] = (unsigned short)(p >> 16);
    }
    v16bf pbf = as_bf16x16(pb);

    // ---- acc += V^T_tile @ P^T ----
#pragma unroll
    for (int f = 0; f < 4; ++f) {
      const unsigned short* varow = vrow[f] + j0;
      v16u vf;
#pragma unroll
      for (int i = 0; i < 8; ++i) {
        int ka = half * 16 + 2 * i;
        unsigned int p = *(const unsigned int*)(varow + ka);
        vf[2 * i]     = (unsigned short)(p & 0xFFFFu);
        vf[2 * i + 1] = (unsigned short)(p >> 16);
      }
      acc[f] = __builtin_amdgcn_wmma_f32_16x16x32_bf16(
          false, as_bf16x16(vf), false, pbf, (short)0, acc[f], false, false);
    }
  };

  // Prologue: stage the first K tile, then run the pipelined loop.
  issueK(0);
  const int dlimit = qt * 16;  // below-diagonal bound: no causal tests needed
  int j0 = 0;
  for (; j0 + 32 <= dlimit; j0 += 32) step(CFalse{}, j0);
  // Exactly one diagonal iteration (covers remaining 16 or 32 keys).
  step(CTrue{}, j0);

  // ---- normalize; lane owns query row -> 8 consecutive dims per f ----
  float inv = 1.0f / rl;
  float* obase = out + (size_t)(bT + qt * 16 + mn) * D + h * 64 + 8 * half;
#pragma unroll
  for (int f = 0; f < 4; ++f) {
    float4 o0 = {acc[f][0] * inv, acc[f][1] * inv, acc[f][2] * inv, acc[f][3] * inv};
    float4 o1 = {acc[f][4] * inv, acc[f][5] * inv, acc[f][6] * inv, acc[f][7] * inv};
    *(float4*)(obase + f * 16)     = o0;
    *(float4*)(obase + f * 16 + 4) = o1;
  }
}

extern "C" void kernel_launch(void* const* d_in, const int* in_sizes, int n_in,
                              void* d_out, int out_size, void* d_ws, size_t ws_size,
                              hipStream_t stream) {
  (void)in_sizes; (void)n_in; (void)out_size; (void)ws_size;
  const float* x   = (const float*)d_in[0];
  const int*   msk = (const int*)d_in[1];
  const float* w_q = (const float*)d_in[2];
  const float* w_d = (const float*)d_in[3];
  const float* w_k = (const float*)d_in[4];
  const float* w_v = (const float*)d_in[5];
  float* out = (float*)d_out;

  const int B = 2, T = 2048, D = 1024, L = 128;
  const int BT = B * T;  // 4096 rows

  unsigned short* ws = (unsigned short*)d_ws;
  size_t o = 0;
  unsigned short* xb  = ws + o; o += (size_t)BT * D;  // x      [BT,D]
  unsigned short* wqT = ws + o; o += (size_t)D * D;   // w_q^T
  unsigned short* wdT = ws + o; o += (size_t)D * L;   // w_d^T
  unsigned short* wkT = ws + o; o += (size_t)L * D;   // w_k^T
  unsigned short* wvT = ws + o; o += (size_t)L * D;   // w_v^T
  unsigned short* qb  = ws + o; o += (size_t)BT * D;  // q      [BT,D]
  unsigned short* lb  = ws + o; o += (size_t)BT * L;  // latent [BT,L]
  unsigned short* kbm = ws + o; o += (size_t)BT * D;  // k      [BT,D]
  unsigned short* vtb = ws + o; o += (size_t)BT * D;  // v^T    [D,BT]
  float* mbf = (float*)(ws + o); o += (size_t)BT * 2; // mask bias [BT] f32

  cvt_f32_bf16<<<2048, 256, 0, stream>>>(x, xb, BT * D);
  cvt_f32_bf16_t<<<1024, 256, 0, stream>>>(w_q, wqT, D, D);
  cvt_f32_bf16_t<<<512,  256, 0, stream>>>(w_d, wdT, D, L);
  cvt_f32_bf16_t<<<512,  256, 0, stream>>>(w_k, wkT, L, D);
  cvt_f32_bf16_t<<<512,  256, 0, stream>>>(w_v, wvT, L, D);
  mask_bias<<<16, 256, 0, stream>>>(msk, mbf, BT);

  dim3 blk(128);
  gemm_bf16<false><<<dim3(D / 128, BT / 64), blk, 0, stream>>>(xb, wqT, qb, BT, D, D);
  gemm_bf16<false><<<dim3(L / 128, BT / 64), blk, 0, stream>>>(xb, wdT, lb, BT, L, D);
  gemm_bf16<false><<<dim3(D / 128, BT / 64), blk, 0, stream>>>(lb, wkT, kbm, BT, D, L);
  gemm_bf16<true><<<dim3(D / 128, BT / 64), blk, 0, stream>>>(lb, wvT, vtb, BT, D, L);

  mla_attn<<<(B * NH * (T / 16)) / 4, 128, 0, stream>>>(qb, kbm, vtb, mbf, out,
                                                        T, D, BT);
}
